// T_21345987461592
// MI455X (gfx1250) — matrix operand order repaired and checked by reading, not compiled, not run on hardware
//
#include <hip/hip_runtime.h>
#include <hip/hip_bf16.h>

// ---- types ----
typedef _Float16 h8  __attribute__((ext_vector_type(8)));
typedef _Float16 h16 __attribute__((ext_vector_type(16)));
typedef float    f8  __attribute__((ext_vector_type(8)));

// Problem constants
#define NB 32
#define NC 16
#define NF 128
#define NT 2048
// Tile config
#define FT 8            // output f-rows per workgroup (== waves per block)
#define TT 128          // t positions per workgroup
#define FROWS (FT + 4)  // 12 input rows incl. halo
#define TCOLS (TT + 4)  // 132 input cols incl. halo
#define CPAD 24         // c-dim padded to 24 halves (48B) -> aligned + conflict-free b128
#define NCHUNK 13       // K = 13 * 32 = 416 (400 real + 16 zero pad)
// ws layout: [0, 13312) A-fragments (f16, fragment-order), [13312, 13376) bias f32
#define WS_BIAS_OFF 13312

// Per-tap LDS byte offset: tap p -> ((p/5)*TCOLS + p%5) * CPAD * 2 bytes.
// Tap 25 is the K-padding tap: its A-fragment entries are exactly 0, so B may
// load any finite in-range data -> alias it to tap 24 (branchless, no divergence).
constexpr int tap_off(int p) {
  return ((p / 5) * TCOLS + (p % 5)) * CPAD * 2;
}
constexpr int OFFB[26] = {
    tap_off(0),  tap_off(1),  tap_off(2),  tap_off(3),  tap_off(4),
    tap_off(5),  tap_off(6),  tap_off(7),  tap_off(8),  tap_off(9),
    tap_off(10), tap_off(11), tap_off(12), tap_off(13), tap_off(14),
    tap_off(15), tap_off(16), tap_off(17), tap_off(18), tap_off(19),
    tap_off(20), tap_off(21), tap_off(22), tap_off(23), tap_off(24),
    tap_off(24)  // p==25 zero-pad tap aliases tap 24 (A side is zero)
};

// ---------------------------------------------------------------------------
// Prep kernel: build W_eff directly in WMMA A-fragment lane order (f16) + bias.
// K index = 32*q + klocal ; tap p = K>>4 (0..24 real, 25 => zero), c = K&15.
// A 16x32 f16 layout as used here (contiguous k, B matches): lane l (M=l&15, hi=l>>4), half i:
//   K = 16*hi + i
// ---------------------------------------------------------------------------
__global__ void weff_prep_kernel(const float* __restrict__ w1, const float* __restrict__ b1,
                                 const float* __restrict__ w3, const float* __restrict__ b3,
                                 const float* __restrict__ w5, const float* __restrict__ b5,
                                 const float* __restrict__ wf, const float* __restrict__ bf,
                                 _Float16* __restrict__ wsA, float* __restrict__ wsB) {
  const int q = blockIdx.x;
  const int l = threadIdx.x;  // 0..31
  if (q == NCHUNK) {          // bias block
    if (l < NC) {
      float acc = bf[l];
      for (int m = 0; m < 16; ++m) {
        acc += wf[l * 48 + m] * b1[m];
        acc += wf[l * 48 + 16 + m] * b3[m];
        acc += wf[l * 48 + 32 + m] * b5[m];
      }
      wsB[l] = acc;
    }
    return;
  }
  const int hi = l >> 4;
  const int M = l & 15;
  for (int i = 0; i < 16; ++i) {
    const int klocal = 16 * hi + i;
    const int K = 32 * q + klocal;
    const int p = K >> 4;  // tap
    const int c = K & 15;  // input channel
    float acc = 0.f;
    if (p < 25) {
      const int ky = p / 5, kx = p % 5;
      for (int m = 0; m < 16; ++m) {
        float v = wf[M * 48 + 32 + m] * w5[((m * 16 + c) * 5 + ky) * 5 + kx];
        if (ky >= 1 && ky <= 3 && kx >= 1 && kx <= 3)
          v += wf[M * 48 + 16 + m] * w3[((m * 16 + c) * 3 + (ky - 1)) * 3 + (kx - 1)];
        if (ky == 2 && kx == 2)
          v += wf[M * 48 + m] * w1[m * 16 + c];
        acc += v;
      }
    }
    wsA[q * 512 + l * 16 + i] = (_Float16)acc;  // zero for the K-pad tap
  }
}

// swap-permutation scatter target for channel M at source position ty (ty%16==n)
__device__ __forceinline__ unsigned perm_dest(int M, int n, unsigned ty) {
  if (n == M) {
    if (ty < NT - 2) return ty + 1;
  } else if (n == ((M + 1) & 15)) {
    if (ty >= 1 && ty <= NT - 2) return ty - 1;
  }
  return ty;
}

// ---------------------------------------------------------------------------
// Fused 5x5 conv via WMMA f32<-f16. Grid: (T/TT, F/FT, NB), block 256 (8 waves).
// ---------------------------------------------------------------------------
__global__ __launch_bounds__(256) void fused_conv_kernel(
    const float* __restrict__ x, const _Float16* __restrict__ wsA,
    const float* __restrict__ wsB, float* __restrict__ out) {
  __shared__ _Float16 xs[FROWS * TCOLS * CPAD];  // 76032 B

  const int tid = threadIdx.x;
  const int b = blockIdx.z;
  const int f0 = blockIdx.y * FT;
  const int t0 = blockIdx.x * TT;

  // --- stage x tile into LDS as f16, layout [fr][tc][c], zero-padded halo ---
  const int NE = NC * FROWS * TCOLS;  // 25344
  for (int e = tid; e < NE; e += 256) {
    const int c = e / (FROWS * TCOLS);
    const int rem = e - c * (FROWS * TCOLS);
    const int fr = rem / TCOLS;
    const int tc = rem - fr * TCOLS;
    const int fg = f0 - 2 + fr;
    const int tg = t0 - 2 + tc;
    float v = 0.f;
    if (fg >= 0 && fg < NF && tg >= 0 && tg < NT)
      v = x[(size_t)((b * NC + c) * NF + fg) * NT + tg];
    xs[(fr * TCOLS + tc) * CPAD + c] = (_Float16)v;
  }
  __syncthreads();

  const int w = tid >> 5;    // wave id == output f-row within tile
  const int lane = tid & 31;
  const int hi = lane >> 4;
  const int n = lane & 15;
  const int f = f0 + w;

  float bias[8];
#pragma unroll
  for (int r = 0; r < 8; ++r) bias[r] = wsB[r + 8 * hi];

  // --- hoist all A-fragments into registers (13 x 8 VGPRs) ---
  const char* aB = (const char*)wsA + lane * 32;
  h16 afrag[NCHUNK];
#pragma unroll
  for (int q = 0; q < NCHUNK; ++q) {
    const h8 alo = *(const h8*)(aB + q * 1024);
    const h8 ahi = *(const h8*)(aB + q * 1024 + 16);
    afrag[q] = __builtin_shufflevector(alo, ahi, 0, 1, 2, 3, 4, 5, 6, 7, 8, 9, 10,
                                       11, 12, 13, 14, 15);
  }

  // loop-invariant 32-bit output base (element index), channel block = 8*hi
  const unsigned obase = ((unsigned)(b * NC + 8 * hi) * NF + (unsigned)f) * NT;
  const char* xbase = (const char*)xs + (unsigned)(w * TCOLS + n) * (CPAD * 2);

  struct B2 { h16 t0, t1; };

#pragma unroll 1
  for (int pr = 0; pr < TT / 32; ++pr) {
    const unsigned tl0 = (unsigned)pr * 32u;
    const char* xb0 = xbase + tl0 * (CPAD * 2);

    // B-fragment loader for chunk q (both tiles of the pair)
    auto loadB = [&](int q) -> B2 {
      const int off = hi ? OFFB[2 * q + 1] : OFFB[2 * q];
      const char* pb = xb0 + off;
      const h8 l0 = *(const h8*)(pb);
      const h8 h0 = *(const h8*)(pb + 16);
      const h8 l1 = *(const h8*)(pb + 16 * CPAD * 2);       // +768
      const h8 h1 = *(const h8*)(pb + 16 * CPAD * 2 + 16);  // +784
      B2 r;
      r.t0 = __builtin_shufflevector(l0, h0, 0, 1, 2, 3, 4, 5, 6, 7, 8, 9, 10, 11,
                                     12, 13, 14, 15);
      r.t1 = __builtin_shufflevector(l1, h1, 0, 1, 2, 3, 4, 5, 6, 7, 8, 9, 10, 11,
                                     12, 13, 14, 15);
      return r;
    };

    f8 acc0 = {};
    f8 acc1 = {};
    B2 cur = loadB(0);
#pragma unroll
    for (int q = 0; q < NCHUNK; ++q) {
      B2 nxt = cur;
      if (q + 1 < NCHUNK) nxt = loadB(q + 1);  // prefetch next chunk's B
      acc0 = __builtin_amdgcn_wmma_f32_16x16x32_f16(false, afrag[q], false, cur.t0,
                                                    (short)0, acc0, false, false);
      acc1 = __builtin_amdgcn_wmma_f32_16x16x32_f16(false, afrag[q], false, cur.t1,
                                                    (short)0, acc1, false, false);
      cur = nxt;
    }

    // --- bias + swap-permutation scatter store (32-bit GVS addressing) ---
    const unsigned ty0 = (unsigned)t0 + tl0 + (unsigned)n;
#pragma unroll
    for (int r = 0; r < 8; ++r) {
      const int M = r + 8 * hi;
      const unsigned roff = obase + (unsigned)r * (NF * NT);  // +r*1MB folds to imm
      out[roff + perm_dest(M, n, ty0)] = acc0[r] + bias[r];
      out[roff + perm_dest(M, n, ty0 + 16u)] = acc1[r] + bias[r];
    }
  }
}

extern "C" void kernel_launch(void* const* d_in, const int* in_sizes, int n_in,
                              void* d_out, int out_size, void* d_ws, size_t ws_size,
                              hipStream_t stream) {
  const float* x = (const float*)d_in[0];
  const float* w1 = (const float*)d_in[1];
  const float* b1 = (const float*)d_in[2];
  const float* w3 = (const float*)d_in[3];
  const float* b3 = (const float*)d_in[4];
  const float* w5 = (const float*)d_in[5];
  const float* b5 = (const float*)d_in[6];
  const float* wf = (const float*)d_in[7];
  const float* bf = (const float*)d_in[8];

  _Float16* wsA = (_Float16*)d_ws;
  float* wsB = (float*)((char*)d_ws + WS_BIAS_OFF);

  weff_prep_kernel<<<dim3(NCHUNK + 1), dim3(32), 0, stream>>>(w1, b1, w3, b3, w5, b5,
                                                              wf, bf, wsA, wsB);
  fused_conv_kernel<<<dim3(NT / TT, NF / FT, NB), dim3(256), 0, stream>>>(
      x, wsA, wsB, (float*)d_out);
}
